// DeeperGCN_19464791785732
// MI455X (gfx1250) — matrix operand order repaired and checked by reading, not compiled, not run on hardware
//
#include <hip/hip_runtime.h>
#include <hip/hip_bf16.h>
#include <math.h>

#define HID  128
#define HID2 256
#define EPS_LN  1e-5f
#define EPS_MSG 1e-7f

typedef __attribute__((ext_vector_type(16))) _Float16 v16h;
typedef __attribute__((ext_vector_type(8)))  _Float16 v8h;
typedef __attribute__((ext_vector_type(8)))  float    v8f;

union AFrag { v16h v; v8h h[2]; };

// ---- WMMA fragment helpers -------------------------------------------------
// A (16x32 f16): lane L<16 holds row M=L, K = k0+0..7 and k0+16..23;
// lane L>=16 holds row M=L-16, K = k0+8..15 and k0+24..31.
__device__ __forceinline__ v16h load_a_frag(const _Float16* A, int lda,
                                            int row0, int k0, int lane) {
  const int m  = row0 + (lane & 15);
  const int kb = k0 + ((lane >> 4) << 3);
  AFrag u;
  u.h[0] = *(const v8h*)(A + (size_t)m * lda + kb);
  u.h[1] = *(const v8h*)(A + (size_t)m * lda + kb + 16);
  return u.v;
}

// B (32x16 f16) pre-packed: tile buffer [tile][lane][16 halves] contiguous.
__device__ __forceinline__ v16h load_b_frag(const _Float16* Bp, int tile, int lane) {
  return *(const v16h*)(Bp + (((size_t)tile * 32 + lane) << 4));
}

__device__ __forceinline__ v8f wmma_f16(v16h a, v16h b, v8f c) {
  return __builtin_amdgcn_wmma_f32_16x16x32_f16(false, a, false, b, (short)0, c,
                                                false, false);
}

// ---- gfx1250 async global->LDS (ASYNCcnt path) -----------------------------
__device__ __forceinline__ void async_ld_b128(unsigned lds_off, const void* gptr) {
  asm volatile("global_load_async_to_lds_b128 %0, %1, off"
               :: "v"(lds_off), "v"(gptr) : "memory");
}
__device__ __forceinline__ void wait_async0() {
  asm volatile("s_wait_asynccnt 0" ::: "memory");
}

// ---- small utility kernels -------------------------------------------------
__global__ void k_zero_f32(float* p, int n) {
  int i = blockIdx.x * blockDim.x + threadIdx.x;
  if (i < n) p[i] = 0.f;
}
__global__ void k_zero_i32(int* p, int n) {
  int i = blockIdx.x * blockDim.x + threadIdx.x;
  if (i < n) p[i] = 0;
}
__global__ void k_zero_f16(_Float16* p, int n) {
  int i = blockIdx.x * blockDim.x + threadIdx.x;
  if (i < n) p[i] = (_Float16)0.f;
}
__global__ void k_cast_f16(const float* s, _Float16* d, int n) {
  int i = blockIdx.x * blockDim.x + threadIdx.x;
  if (i < n) d[i] = (_Float16)s[i];
}

// Pack a row-major f32 [K x Nc] weight into f16 B-fragment tiles.
__global__ void k_pack_b(const float* W, _Float16* Bp, int K, int Nc) {
  int idx = blockIdx.x * blockDim.x + threadIdx.x;
  if (idx >= K * Nc) return;
  int e    = idx & 15;
  int lane = (idx >> 4) & 31;
  int tile = idx >> 9;            // 512 halves per tile
  int NT   = Nc >> 4;
  int kt   = tile / NT;
  int nt   = tile - kt * NT;
  int k = kt * 32 + ((lane >> 4) << 3) + (e & 7) + ((e >> 3) << 4);
  int n = nt * 16 + (lane & 15);
  Bp[idx] = (_Float16)W[(size_t)k * Nc + n];
}

// ---- CSR build -------------------------------------------------------------
__global__ void k_count_deg(const int* ei, int E, int* deg) {
  int e = blockIdx.x * blockDim.x + threadIdx.x;
  if (e < E) atomicAdd(&deg[ei[E + e]], 1);
}
__global__ void k_scan(const int* deg, int* rowptr, int N, int E) {
  __shared__ int part[1024];
  int t = threadIdx.x;
  int chunk = (N + 1023) >> 10;
  int beg = t * chunk;
  int end = beg + chunk; if (end > N) end = N;
  int s = 0;
  for (int i = beg; i < end; ++i) s += deg[i];
  part[t] = s;
  __syncthreads();
  for (int off = 1; off < 1024; off <<= 1) {
    int v = (t >= off) ? part[t - off] : 0;
    __syncthreads();
    part[t] += v;
    __syncthreads();
  }
  int run = (t == 0) ? 0 : part[t - 1];
  for (int i = beg; i < end; ++i) { rowptr[i] = run; run += deg[i]; }
  if (t == 1023) rowptr[N] = part[1023];
}
__global__ void k_fill_csr(const int* ei, int E, const int* rowptr,
                           int* cursor, int* srcidx) {
  int e = blockIdx.x * blockDim.x + threadIdx.x;
  if (e >= E) return;
  int s = ei[e], d = ei[E + e];
  int pos = rowptr[d] + atomicAdd(&cursor[d], 1);
  srcidx[pos] = s;
}

// ---- block reductions (128 threads) ---------------------------------------
__device__ __forceinline__ float blk_sum_128(float v, float* red, int f) {
  red[f] = v; __syncthreads();
  #pragma unroll
  for (int s = 64; s > 0; s >>= 1) {
    if (f < s) red[f] += red[f + s];
    __syncthreads();
  }
  float r = red[0];
  __syncthreads();
  return r;
}

// ---- pre-layer: r = leaky_relu(LN(h)) ; rnorm = ||r|| ----------------------
__global__ void k_pre_norm(const float* __restrict__ h, const float* g,
                           const float* b, float* r, float* rnorm) {
  __shared__ float red[HID];
  int v = blockIdx.x, f = threadIdx.x;
  float x  = h[(size_t)v * HID + f];
  float mu = blk_sum_128(x, red, f) * (1.f / HID);
  float d  = x - mu;
  float var = blk_sum_128(d * d, red, f) * (1.f / HID);
  float y  = d * rsqrtf(var + EPS_LN) * g[f] + b[f];
  float rv = y > 0.f ? y : 0.01f * y;
  r[(size_t)v * HID + f] = rv;
  float n2 = blk_sum_128(rv * rv, red, f);
  if (f == 0) rnorm[v] = sqrtf(n2);
}

// ---- per-node ONLINE-softmax aggregation + MessageNorm ---------------------
// Single pass over incoming edges (running max with rescaled sums) halves the
// dominant gather traffic vs the max/exp two-pass formulation.
__global__ void k_node_agg(const float* __restrict__ r, const float* rnorm,
                           const int* rowptr, const int* srcidx,
                           const float* tarr, const float* scarr, int layer,
                           _Float16* out16) {
  __shared__ float red[HID];
  int v = blockIdx.x, f = threadIdx.x;
  float tc = tarr[layer], sc = scarr[layer];
  int beg = rowptr[v], end = rowptr[v + 1];
  float M = -INFINITY, ssum = 0.f, wsum = 0.f;
  for (int e = beg; e < end; ++e) {
    if (e + 1 < end)
      __builtin_prefetch(&r[(size_t)srcidx[e + 1] * HID + f], 0, 0);
    float m  = fmaxf(r[(size_t)srcidx[e] * HID + f], 0.f) + EPS_MSG;
    float z  = m * tc;
    float nm = fmaxf(M, z);
    float so = expf(M - nm);     // first iter: expf(-inf)=0
    float ee = expf(z - nm);
    ssum = ssum * so + ee;
    wsum = wsum * so + m * ee;
    M = nm;
  }
  float agg = wsum / (ssum + 1e-16f);   // no edges -> 0/1e-16 = 0
  float n2  = blk_sum_128(agg * agg, red, f);
  float inv = rnorm[v] * sc / fmaxf(sqrtf(n2), 1e-12f);
  float out = r[(size_t)v * HID + f] + agg * inv;
  out16[(size_t)v * HID + f] = (_Float16)out;
}

// ---- encoder GEMM: h = x16 @ encWp + bias (K=128, Nc=128, 1 tile/wave) -----
__global__ void k_enc_gemm(const _Float16* __restrict__ A,
                           const _Float16* __restrict__ Bp,
                           const float* bias, float* C, int M) {
  int wave = (blockIdx.x * blockDim.x + threadIdx.x) >> 5;
  int lane = threadIdx.x & 31;
  const int NT = HID / 16;                 // 8
  int mt = wave / NT, nt = wave - mt * NT;
  if (mt * 16 >= M) return;                // wave-uniform
  v16h a[4], b[4];
  #pragma unroll
  for (int kt = 0; kt < 4; ++kt) {         // preload -> overlapped loads
    a[kt] = load_a_frag(A, HID, mt * 16, kt * 32, lane);
    b[kt] = load_b_frag(Bp, kt * NT + nt, lane);
  }
  v8f acc = {};
  #pragma unroll
  for (int kt = 0; kt < 4; ++kt) acc = wmma_f16(a[kt], b[kt], acc);
  int col   = nt * 16 + (lane & 15);
  int rbase = mt * 16 + ((lane >> 4) << 3);
  float bv  = bias[col];
  #pragma unroll
  for (int j = 0; j < 8; ++j)
    C[(size_t)(rbase + j) * HID + col] = acc[j] + bv;
}

// ---- fused MLP: h += relu(LN(out16 @ W1 + b1)) @ W2 + b2  (16 rows/block) --
__global__ __launch_bounds__(256)
void k_mlp_fused(const _Float16* __restrict__ out16,
                 const _Float16* __restrict__ W1p, const float* b1v,
                 const float* mg, const float* mb,
                 const _Float16* __restrict__ W2p, const float* b2v,
                 float* __restrict__ h, int Nrows) {
  __shared__ __align__(32) _Float16 astage[16 * HID];   // 4KB A tile
  __shared__ __align__(32) float    zbuf[16 * HID2];
  __shared__ __align__(32) _Float16 abuf[16 * HID2];
  __shared__ float ps[16][16], ps2[16][16], mu_s[16], rs_s[16];
  int mt = blockIdx.x, tid = threadIdx.x;
  int wave = tid >> 5, lane = tid & 31;
  int cl = lane & 15, rj = (lane >> 4) << 3;

  // async-stage the 16x128 f16 A tile into LDS (one b128 per thread)
  {
    unsigned lds_off = ((unsigned)(uintptr_t)astage) + (unsigned)tid * 16u;
    const _Float16* g = out16 + (size_t)mt * 16 * HID + (size_t)tid * 8;
    async_ld_b128(lds_off, g);
    wait_async0();
  }
  __syncthreads();

  // GEMM1: 256 cols = 16 n-tiles; wave w owns tiles 2w, 2w+1; K=128 -> 4 steps
  v16h af1[4], b0[4], b1f[4];
  #pragma unroll
  for (int kt = 0; kt < 4; ++kt) {
    af1[kt] = load_a_frag(astage, HID, 0, kt * 32, lane);
    b0[kt]  = load_b_frag(W1p, kt * 16 + 2 * wave,     lane);
    b1f[kt] = load_b_frag(W1p, kt * 16 + 2 * wave + 1, lane);
  }
  v8f a0 = {}, a1 = {};
  #pragma unroll
  for (int kt = 0; kt < 4; ++kt) {
    a0 = wmma_f16(af1[kt], b0[kt],  a0);
    a1 = wmma_f16(af1[kt], b1f[kt], a1);
  }
  #pragma unroll
  for (int j = 0; j < 8; ++j) {
    int c0 = (2 * wave) * 16 + cl, c1 = c0 + 16;
    zbuf[(rj + j) * HID2 + c0] = a0[j] + b1v[c0];
    zbuf[(rj + j) * HID2 + c1] = a1[j] + b1v[c1];
  }
  __syncthreads();

  // LayerNorm over 256 per row, ReLU, convert to f16 A for GEMM2
  int row = tid >> 4, lr = tid & 15;
  float s = 0.f, s2 = 0.f;
  #pragma unroll
  for (int i = 0; i < 16; ++i) {
    float z = zbuf[row * HID2 + lr + 16 * i];
    s += z; s2 += z * z;
  }
  ps[row][lr] = s; ps2[row][lr] = s2;
  __syncthreads();
  if (lr == 0) {
    float S = 0.f, S2 = 0.f;
    #pragma unroll
    for (int i = 0; i < 16; ++i) { S += ps[row][i]; S2 += ps2[row][i]; }
    float mu = S * (1.f / HID2);
    mu_s[row] = mu;
    rs_s[row] = rsqrtf(S2 * (1.f / HID2) - mu * mu + EPS_LN);
  }
  __syncthreads();
  float mu = mu_s[row], rs = rs_s[row];
  #pragma unroll
  for (int i = 0; i < 16; ++i) {
    int c = lr + 16 * i;
    float z = (zbuf[row * HID2 + c] - mu) * rs * mg[c] + mb[c];
    abuf[row * HID2 + c] = (_Float16)fmaxf(z, 0.f);
  }
  __syncthreads();

  // GEMM2: [16x256] @ [256x128]; wave w owns n-tile w; K=256 -> 8 steps
  v16h b2f[8];
  #pragma unroll
  for (int kt = 0; kt < 8; ++kt) b2f[kt] = load_b_frag(W2p, kt * 8 + wave, lane);
  v8f acc = {};
  #pragma unroll
  for (int kt = 0; kt < 8; ++kt) {
    v16h af = load_a_frag(abuf, HID2, 0, kt * 32, lane);
    acc = wmma_f16(af, b2f[kt], acc);
  }
  int col = wave * 16 + cl;
  float bv = b2v[col];
  #pragma unroll
  for (int j = 0; j < 8; ++j) {
    int rrow = mt * 16 + rj + j;
    if (rrow < Nrows) h[(size_t)rrow * HID + col] += acc[j] + bv;
  }
}

// ---- final LN + leaky + mean pool ------------------------------------------
__global__ void k_finalize(const float* __restrict__ h, const float* g,
                           const float* b, const int* batch,
                           float* gf, float* cnt) {
  __shared__ float red[HID];
  int v = blockIdx.x, f = threadIdx.x;
  float x  = h[(size_t)v * HID + f];
  float mu = blk_sum_128(x, red, f) * (1.f / HID);
  float d  = x - mu;
  float var = blk_sum_128(d * d, red, f) * (1.f / HID);
  float y  = d * rsqrtf(var + EPS_LN) * g[f] + b[f];
  float lv = y > 0.f ? y : 0.01f * y;
  int gidx = batch[v];
  if (f == 0) atomicAdd(&cnt[gidx], 1.f);
  atomicAdd(&gf[(size_t)gidx * HID + f], lv);
}
__global__ void k_pool_div(float* gf, const float* cnt, int total) {
  int i = blockIdx.x * blockDim.x + threadIdx.x;
  if (i < total) gf[i] /= fmaxf(cnt[i >> 7], 1.f);
}

// ---------------------------------------------------------------------------
extern "C" void kernel_launch(void* const* d_in, const int* in_sizes, int n_in,
                              void* d_out, int out_size, void* d_ws, size_t ws_size,
                              hipStream_t stream) {
  const float* x     = (const float*)d_in[0];
  const int*   ei    = (const int*)d_in[1];
  const int*   batch = (const int*)d_in[2];
  const float* encW  = (const float*)d_in[3];
  const float* encB  = (const float*)d_in[4];
  const float* ln_g  = (const float*)d_in[5];
  const float* ln_b  = (const float*)d_in[6];
  const float* tarr  = (const float*)d_in[7];
  const float* scarr = (const float*)d_in[8];
  const float* W1    = (const float*)d_in[9];
  const float* b1    = (const float*)d_in[10];
  const float* mg    = (const float*)d_in[11];
  const float* mb    = (const float*)d_in[12];
  const float* W2    = (const float*)d_in[13];
  const float* b2    = (const float*)d_in[14];
  const float* fn_g  = (const float*)d_in[15];
  const float* fn_b  = (const float*)d_in[16];

  const int N = in_sizes[0] / HID;
  const int E = in_sizes[1] / 2;
  const int L = in_sizes[7];
  const int G = out_size / HID;
  const int Mpad = (N + 15) & ~15;

  // ---- workspace carve (256B aligned) ----
  char* p = (char*)d_ws;
  auto carve = [&](size_t bytes) {
    void* r = (void*)p;
    p += (bytes + 255) & ~(size_t)255;
    return r;
  };
  _Float16* x16   = (_Float16*)carve((size_t)Mpad * HID * 2);
  _Float16* out16 = (_Float16*)carve((size_t)Mpad * HID * 2);
  _Float16* encWp = (_Float16*)carve((size_t)HID * HID * 2);
  _Float16* W1p   = (_Float16*)carve((size_t)L * HID * HID2 * 2);
  _Float16* W2p   = (_Float16*)carve((size_t)L * HID2 * HID * 2);
  float* h     = (float*)carve((size_t)Mpad * HID * 4);
  float* r     = (float*)carve((size_t)Mpad * HID * 4);
  float* rnorm = (float*)carve((size_t)N * 4);
  int* deg     = (int*)carve((size_t)N * 4);
  int* cursor  = (int*)carve((size_t)N * 4);
  int* rowptr  = (int*)carve((size_t)(N + 1) * 4);
  int* srcidx  = (int*)carve((size_t)E * 4);
  float* cnt   = (float*)carve((size_t)G * 4);
  float* gf    = (float*)d_out;

  auto gs = [](int n) { return (n + 255) / 256; };

  // activations -> f16 (pad rows zeroed)
  k_cast_f16<<<gs(N * HID), 256, 0, stream>>>(x, x16, N * HID);
  if (Mpad > N) {
    int padn = (Mpad - N) * HID;
    k_zero_f16<<<gs(padn), 256, 0, stream>>>(x16 + (size_t)N * HID, padn);
    k_zero_f16<<<gs(padn), 256, 0, stream>>>(out16 + (size_t)N * HID, padn);
  }

  // weight packing (once per launch)
  k_pack_b<<<gs(HID * HID), 256, 0, stream>>>(encW, encWp, HID, HID);
  for (int l = 0; l < L; ++l) {
    k_pack_b<<<gs(HID * HID2), 256, 0, stream>>>(
        W1 + (size_t)l * HID * HID2, W1p + (size_t)l * HID * HID2, HID, HID2);
    k_pack_b<<<gs(HID2 * HID), 256, 0, stream>>>(
        W2 + (size_t)l * HID2 * HID, W2p + (size_t)l * HID2 * HID, HID2, HID);
  }

  // CSR build
  k_zero_i32<<<gs(N), 256, 0, stream>>>(deg, N);
  k_zero_i32<<<gs(N), 256, 0, stream>>>(cursor, N);
  k_count_deg<<<gs(E), 256, 0, stream>>>(ei, E, deg);
  k_scan<<<1, 1024, 0, stream>>>(deg, rowptr, N, E);
  k_fill_csr<<<gs(E), 256, 0, stream>>>(ei, E, rowptr, cursor, srcidx);

  // encoder: h = x @ enc_W + enc_b   (one wave per 16x16 tile)
  {
    int waves = (Mpad / 16) * (HID / 16);
    k_enc_gemm<<<(waves + 3) / 4, 128, 0, stream>>>(x16, encWp, encB, h, Mpad);
  }

  // DeepGCN layers
  for (int l = 0; l < L; ++l) {
    k_pre_norm<<<N, HID, 0, stream>>>(h, ln_g + (size_t)l * HID,
                                      ln_b + (size_t)l * HID, r, rnorm);
    k_node_agg<<<N, HID, 0, stream>>>(r, rnorm, rowptr, srcidx,
                                      tarr, scarr, l, out16);
    k_mlp_fused<<<Mpad / 16, 256, 0, stream>>>(
        out16, W1p + (size_t)l * HID * HID2, b1 + (size_t)l * HID2,
        mg + (size_t)l * HID2, mb + (size_t)l * HID2,
        W2p + (size_t)l * HID2 * HID, b2 + (size_t)l * HID, h, N);
  }

  // final norm + mean pool
  k_zero_f32<<<gs(G * HID), 256, 0, stream>>>(gf, G * HID);
  k_zero_f32<<<gs(G), 256, 0, stream>>>(cnt, G);
  k_finalize<<<N, HID, 0, stream>>>(h, fn_g, fn_b, batch, gf, cnt);
  k_pool_div<<<gs(G * HID), 256, 0, stream>>>(gf, cnt, G * HID);
}
